// HeteroGCN_11699490914986
// MI455X (gfx1250) — matrix operand order repaired and checked by reading, not compiled, not run on hardware
//
#include <hip/hip_runtime.h>
#include <hip/hip_bf16.h>

// ---------------------------------------------------------------------------
// Types for CDNA5 WMMA (gfx1250, wave32)
// ---------------------------------------------------------------------------
typedef __attribute__((ext_vector_type(16))) __bf16 bf16x16;
typedef __attribute__((ext_vector_type(8)))  float  floatx8;

union FragBF {
    unsigned int u[8];
    bf16x16      v;
};

__device__ __forceinline__ unsigned int pack_bf16x2(float a, float b) {
    unsigned int ua = __float_as_uint(a);
    unsigned int ub = __float_as_uint(b);
    // round-to-nearest-even fp32 -> bf16
    unsigned int ra = (ua + 0x7FFFu + ((ua >> 16) & 1u)) >> 16;
    unsigned int rb = (ub + 0x7FFFu + ((ub >> 16) & 1u)) >> 16;
    return (ra & 0xFFFFu) | (rb << 16);
}

// ---------------------------------------------------------------------------
// WMMA GEMM: C[M,256] = A[M,K] * B[K,256], fp32 in/out, bf16 compute/f32 acc.
// Tile: BM=64 x BN=256 (full N in one pass -> A streamed exactly once),
// BK=32. 256 threads = 8 waves arranged 2(M) x 4(N); each wave owns a
// 32x64 patch = 2x4 v_wmma_f32_16x16x32_bf16 fragments (8 WMMA / K-step).
// N fixed at 256 (compile-time) so C stores use immediate offsets.
// K must be a multiple of 32 (here 512 or 256).
// ---------------------------------------------------------------------------
#define BM 64
#define BN 256
#define BK 32
#define GN 256   // fixed N
#define LDA 18   // uint32 (bf16-pair) stride per A row  (16 pairs + 2 pad)
#define LDB 18   // uint32 stride per B column (col-major in LDS)

__global__ __launch_bounds__(256) void gemm_bf16_wmma(
    const float* __restrict__ A, const float* __restrict__ B,
    float* __restrict__ C, int M, int K)
{
    __shared__ unsigned int sA[BM * LDA];  //  4608 B
    __shared__ unsigned int sB[BN * LDB];  // 18432 B

    const int tid    = threadIdx.x;
    const int wave   = tid >> 5;
    const int lane   = tid & 31;
    const int waveM  = wave >> 2;          // 0..1
    const int waveN  = wave & 3;           // 0..3
    const int m_tile = blockIdx.y * BM;

    const int frow = lane & 15;            // row (A) / col (B,C) within frag
    const int half = lane >> 4;            // lane half selects K/M groups

    floatx8 acc[2][4];
    #pragma unroll
    for (int i = 0; i < 2; ++i)
        #pragma unroll
        for (int j = 0; j < 4; ++j)
            acc[i][j] = (floatx8){0.f,0.f,0.f,0.f,0.f,0.f,0.f,0.f};

    for (int k0 = 0; k0 < K; k0 += BK) {
        __syncthreads();
        // ---- stage A tile (64 x 32 fp32 -> bf16 pairs), prefetch next ----
        #pragma unroll
        for (int t = 0; t < (BM * (BK / 2)) / 256; ++t) {
            int i   = tid + t * 256;
            int row = i >> 4;              // /16 pairs per row
            int cp  = i & 15;
            int gm  = m_tile + row;
            float f0 = 0.f, f1 = 0.f;
            if (gm < M) {
                const float* p = A + (size_t)gm * K + k0 + cp * 2;
                f0 = p[0]; f1 = p[1];
                if (k0 + BK < K) __builtin_prefetch(p + BK, 0, 1);
            }
            sA[row * LDA + cp] = pack_bf16x2(f0, f1);
        }
        // ---- stage B tile (32 x 256 fp32 -> bf16 pairs, col-major) ----
        #pragma unroll
        for (int t = 0; t < ((BK / 2) * BN) / 256; ++t) {
            int i   = tid + t * 256;
            int col = i & (BN - 1);        // coalesced over N
            int kp  = i >> 8;              // K pair index 0..15
            const float* p = B + (size_t)(k0 + kp * 2) * GN + col;
            sB[col * LDB + kp] = pack_bf16x2(p[0], p[GN]);
        }
        __syncthreads();

        // ---- fragment loads per ISA VGPR layouts ----
        FragBF a0, a1, bfr[4];
        const int ar = waveM * 32;
        const int bc = waveN * 64;
        #pragma unroll
        for (int v = 0; v < 8; ++v) {
            // 16-bit A 16x32: lanes 0-15 K=0-7,16-23 ; lanes 16-31 K=8-15,24-31
            int kpA = (v < 4) ? (4 * half + v) : (8 + 4 * half + (v - 4));
            // 16-bit B 32x16: lanes 0-15 K=0-7,16-23 ; lanes 16-31 K=8-15,24-31
            int kpB = (v < 4) ? (4 * half + v) : (8 + 4 * half + (v - 4));
            a0.u[v] = sA[(ar + frow)      * LDA + kpA];
            a1.u[v] = sA[(ar + 16 + frow) * LDA + kpA];
            #pragma unroll
            for (int j = 0; j < 4; ++j)
                bfr[j].u[v] = sB[(bc + j * 16 + frow) * LDB + kpB];
        }
        #pragma unroll
        for (int j = 0; j < 4; ++j) {
            acc[0][j] = __builtin_amdgcn_wmma_f32_16x16x32_bf16(
                false, a0.v, false, bfr[j].v, (short)0, acc[0][j], false, false);
            acc[1][j] = __builtin_amdgcn_wmma_f32_16x16x32_bf16(
                false, a1.v, false, bfr[j].v, (short)0, acc[1][j], false, false);
        }
    }

    // ---- store C per 16x16 f32 D layout (col=lane%16, row=r+8*half) ----
    const bool full = (m_tile + BM) <= M;   // uniform across block
    #pragma unroll
    for (int i = 0; i < 2; ++i) {
        #pragma unroll
        for (int j = 0; j < 4; ++j) {
            int gm0 = m_tile + waveM * 32 + i * 16 + half * 8;
            int gn  = waveN * 64 + j * 16 + frow;
            float* p = C + (size_t)gm0 * GN + gn;
            if (full) {
                #pragma unroll
                for (int r = 0; r < 8; ++r) p[r * GN] = acc[i][j][r];
            } else {
                #pragma unroll
                for (int r = 0; r < 8; ++r)
                    if (gm0 + r < M) p[r * GN] = acc[i][j][r];
            }
        }
    }
}

// ---------------------------------------------------------------------------
// Edge scatter: out[dst] += hw[src] * inv_s[src] * inv_d[dst]  (256 feats)
// One wave32 per edge, float4 gathers, f32 global atomics for scatter.
// ---------------------------------------------------------------------------
__global__ __launch_bounds__(256) void scatter_gcn(
    const float* __restrict__ hw, const int* __restrict__ src,
    const int* __restrict__ dst, const float* __restrict__ inv_s,
    const float* __restrict__ inv_d, float* __restrict__ out, int E)
{
    int e    = (int)((blockIdx.x * (unsigned)blockDim.x + threadIdx.x) >> 5);
    int lane = threadIdx.x & 31;
    if (e >= E) return;
    int s = src[e], d = dst[e];
    float nrm = inv_s[s] * inv_d[d];
    const float4* hp = (const float4*)(hw + (size_t)s * 256);
    float*        op = out + (size_t)d * 256;
    #pragma unroll
    for (int j = 0; j < 2; ++j) {
        int    idx = lane + 32 * j;
        float4 v   = hp[idx];
        float* o   = op + idx * 4;
        atomicAdd(o + 0, v.x * nrm);
        atomicAdd(o + 1, v.y * nrm);
        atomicAdd(o + 2, v.z * nrm);
        atomicAdd(o + 3, v.w * nrm);
    }
}

__global__ void deg_count(const int* __restrict__ idx, float* __restrict__ deg, int E) {
    int i = blockIdx.x * blockDim.x + threadIdx.x;
    if (i < E) atomicAdd(deg + idx[i], 1.0f);
}

__global__ void inv_sqrt_deg(float* __restrict__ deg, float add, int n) {
    int i = blockIdx.x * blockDim.x + threadIdx.x;
    if (i < n) {
        float d = deg[i] + add;
        deg[i] = (d > 0.f) ? rsqrtf(d) : 0.f;
    }
}

__global__ void fill_f32(float* __restrict__ p, float v, long n) {
    long i = blockIdx.x * (long)blockDim.x + threadIdx.x;
    long s = (long)gridDim.x * blockDim.x;
    for (; i < n; i += s) p[i] = v;
}

// rest nodes: relu(agg + self_term + b_ur + b_rr)
__global__ void rest_finalize(float* __restrict__ agg, const float* __restrict__ hw_self,
                              const float* __restrict__ inv_ns, const float* __restrict__ inv_nd,
                              const float* __restrict__ ba, const float* __restrict__ bb, int n) {
    int i = blockIdx.x * blockDim.x + threadIdx.x;
    if (i >= n) return;
    int node = i >> 8, f = i & 255;
    float v = agg[i] + hw_self[i] * (inv_ns[node] * inv_nd[node]) + ba[f] + bb[f];
    agg[i] = fmaxf(v, 0.f);
}

// user nodes: relu(agg + b)
__global__ void user_finalize(float* __restrict__ u, const float* __restrict__ b, int n) {
    int i = blockIdx.x * blockDim.x + threadIdx.x;
    if (i >= n) return;
    float v = u[i] + b[i & 255];
    u[i] = fmaxf(v, 0.f);
}

// ---------------------------------------------------------------------------
// Host-side orchestration
// ---------------------------------------------------------------------------
static const int NU = 50000, NR = 20000, DIN = 512, H = 256;
static const int EREV = 500000, ENEAR = 250000;

static inline void gemm(const float* A, const float* B, float* C,
                        int M, int K, hipStream_t s) {
    dim3 grid(1, (M + BM - 1) / BM);   // N=256 covered in one tile column
    gemm_bf16_wmma<<<grid, dim3(256), 0, s>>>(A, B, C, M, K);
}

extern "C" void kernel_launch(void* const* d_in, const int* in_sizes, int n_in,
                              void* d_out, int out_size, void* d_ws, size_t ws_size,
                              hipStream_t stream) {
    const float* x_user    = (const float*)d_in[0];
    const float* x_rest    = (const float*)d_in[1];
    const int*   rev_src   = (const int*)d_in[2];
    const int*   rev_dst   = (const int*)d_in[3];
    const int*   near_src  = (const int*)d_in[4];
    const int*   near_dst  = (const int*)d_in[5];
    const float* Win_user  = (const float*)d_in[6];
    const float* Win_rest  = (const float*)d_in[7];
    const float* W1_ur = (const float*)d_in[8];  const float* b1_ur = (const float*)d_in[9];
    const float* W1_ru = (const float*)d_in[10]; const float* b1_ru = (const float*)d_in[11];
    const float* W1_rr = (const float*)d_in[12]; const float* b1_rr = (const float*)d_in[13];
    const float* W2_ur = (const float*)d_in[14]; const float* b2_ur = (const float*)d_in[15];
    const float* W2_ru = (const float*)d_in[16]; const float* b2_ru = (const float*)d_in[17];
    const float* W2_rr = (const float*)d_in[18]; const float* b2_rr = (const float*)d_in[19];
    const float* Wout_user = (const float*)d_in[20];
    const float* Wout_rest = (const float*)d_in[21];

    // workspace carve-up (floats)
    float* U1 = (float*)d_ws;            // user activations   [NU*256]
    float* U2 = U1 + (size_t)NU * 256;   // user GEMM scratch  [NU*256]
    float* R1 = U2 + (size_t)NU * 256;   // rest activations   [NR*256]
    float* R2 = R1 + (size_t)NR * 256;   // rest GEMM scratch  [NR*256]
    float* R3 = R2 + (size_t)NR * 256;   // rest GEMM scratch  [NR*256]
    float* inv_us = R3 + (size_t)NR * 256;  // user deg^-1/2 (rev_src)   [NU]
    float* inv_rd = inv_us + NU;            // rest deg^-1/2 (rev_dst)   [NR]
    float* inv_ns = inv_rd + NR;            // rest deg^-1/2 (near_src+1)[NR]
    float* inv_nd = inv_ns + NR;            // rest deg^-1/2 (near_dst+1)[NR]

    // ---- degrees -> inverse sqrt (self-loop +1 for homogeneous relation) ----
    long ndeg = (long)NU + 3L * NR;
    fill_f32<<<1024, 256, 0, stream>>>(inv_us, 0.f, ndeg);
    deg_count<<<(EREV  + 255) / 256, 256, 0, stream>>>(rev_src,  inv_us, EREV);
    deg_count<<<(EREV  + 255) / 256, 256, 0, stream>>>(rev_dst,  inv_rd, EREV);
    deg_count<<<(ENEAR + 255) / 256, 256, 0, stream>>>(near_src, inv_ns, ENEAR);
    deg_count<<<(ENEAR + 255) / 256, 256, 0, stream>>>(near_dst, inv_nd, ENEAR);
    inv_sqrt_deg<<<(NU + 255) / 256, 256, 0, stream>>>(inv_us, 0.f, NU);
    inv_sqrt_deg<<<(NR + 255) / 256, 256, 0, stream>>>(inv_rd, 0.f, NR);
    inv_sqrt_deg<<<(NR + 255) / 256, 256, 0, stream>>>(inv_ns, 1.f, NR);
    inv_sqrt_deg<<<(NR + 255) / 256, 256, 0, stream>>>(inv_nd, 1.f, NR);

    // ---- input projections ----
    gemm(x_user, Win_user, U1, NU, DIN, stream);
    gemm(x_rest, Win_rest, R1, NR, DIN, stream);

    const float* Wur[2] = {W1_ur, W2_ur}; const float* bur[2] = {b1_ur, b2_ur};
    const float* Wru[2] = {W1_ru, W2_ru}; const float* bru[2] = {b1_ru, b2_ru};
    const float* Wrr[2] = {W1_rr, W2_rr}; const float* brr[2] = {b1_rr, b2_rr};

    for (int l = 0; l < 2; ++l) {
        // relation GEMMs (consume U1/R1)
        gemm(U1, Wur[l], U2, NU, H, stream);   // hw for user->rest
        gemm(R1, Wrr[l], R2, NR, H, stream);   // hw for rest->rest (+ self)
        gemm(R1, Wru[l], R3, NR, H, stream);   // hw for rest->user

        // rest-target aggregation into R1 (old hr now dead)
        fill_f32<<<2048, 256, 0, stream>>>(R1, 0.f, (long)NR * 256);
        scatter_gcn<<<(EREV  + 7) / 8, 256, 0, stream>>>(U2, rev_src,  rev_dst,
                                                         inv_us, inv_rd, R1, EREV);
        scatter_gcn<<<(ENEAR + 7) / 8, 256, 0, stream>>>(R2, near_src, near_dst,
                                                         inv_ns, inv_nd, R1, ENEAR);
        rest_finalize<<<(NR * 256 + 255) / 256, 256, 0, stream>>>(
            R1, R2, inv_ns, inv_nd, bur[l], brr[l], NR * 256);

        // user-target aggregation into U1 (old hu now dead)
        fill_f32<<<4096, 256, 0, stream>>>(U1, 0.f, (long)NU * 256);
        scatter_gcn<<<(EREV + 7) / 8, 256, 0, stream>>>(R3, rev_dst, rev_src,
                                                        inv_rd, inv_us, U1, EREV);
        user_finalize<<<(NU * 256 + 255) / 256, 256, 0, stream>>>(U1, bru[l], NU * 256);
    }

    // ---- output projections straight into d_out ----
    float* out_f = (float*)d_out;
    gemm(U1, Wout_user, out_f,                  NU, H, stream);
    gemm(R1, Wout_rest, out_f + (size_t)NU * H, NR, H, stream);
}
